// AttnBlock_32899449487664
// MI455X (gfx1250) — matrix operand order repaired and checked by reading, not compiled, not run on hardware
//
#include <hip/hip_runtime.h>
#include <hip/hip_bf16.h>

// ---------------------------------------------------------------------------
// AttnBlock for MI455X (gfx1250, wave32, WMMA).
// B=8, H=W=64 -> N=4096 tokens/batch, C=512.
//   h = x/255 ; Q=hWq K=hWk V=hWv ; S=QK^T ; P=softmax(S) ; O=PV ; out=x+OWp
// All GEMMs use v_wmma_f32_16x16x32_bf16 (bf16 in, f32 accum).
// The 16x4096 logits row-block lives entirely in 256KB LDS (CDNA5: 320KB/WG),
// so the 512MB logits tensor never touches HBM.
// Each wave computes a 16x64 output strip. GEMM loops are manually software-
// pipelined AND pinned with sched_group_barrier (loads-then-WMMAs per step)
// so fragment-load latency hides under the matrix ops.
// ---------------------------------------------------------------------------

typedef __attribute__((ext_vector_type(16))) __bf16 v16bf;
typedef __attribute__((ext_vector_type(8)))  __bf16 v8bf;
typedef __attribute__((ext_vector_type(8)))  float  v8f;

static constexpr int C_DIM  = 512;
static constexpr int NTOK   = 4096;            // tokens per batch (64*64)
static constexpr int BATCH  = 8;
static constexpr int TOKENS = BATCH * NTOK;    // 32768

// IGroupLP scheduling groups: 0x8 = MFMA/WMMA, 0x20 = VMEM read, 0x100 = DS read
#if __has_builtin(__builtin_amdgcn_sched_group_barrier)
#define SCHED_GROUP(mask, n) __builtin_amdgcn_sched_group_barrier((mask), (n), 0)
#else
#define SCHED_GROUP(mask, n)
#endif

// ---- fragment helpers (wave32 WMMA VGPR layouts, CDNA5 ISA) -----

// A-matrix 16x32 bf16, row-major source. lane: row = lane&15, kg = lane>>4.
// comps 0..7  -> k = kc + kg*8 + c        (contiguous, 16B)
// comps 8..15 -> k = kc + 16 + kg*8 + c-8 (contiguous, 16B)
__device__ __forceinline__ v16bf load_a_bf16(const __bf16* rowptr, int kc, int kg) {
  const __bf16* p = rowptr + kc + kg * 8;
  v8bf lo = *(const v8bf*)p;
  v8bf hi = *(const v8bf*)(p + 16);
  v16bf a;
#pragma unroll
  for (int i = 0; i < 8; ++i) { a[i] = lo[i]; a[8 + i] = hi[i]; }
  return a;
}

// A-fragment built from f32 LDS (unnormalized softmax numerators).
__device__ __forceinline__ v16bf load_a_f32(const float* rowptr, int kc, int kg) {
  const float* p = rowptr + kc + kg * 8;
  v16bf a;
#pragma unroll
  for (int i = 0; i < 8; ++i) {
    a[i]     = (__bf16)p[i];
    a[8 + i] = (__bf16)p[16 + i];
  }
  return a;
}

// B-matrix 32x16 bf16: lane col n = lane&15, half = lane>>4,
// comps 0..7 -> k = kc + half*8 + c, comps 8..15 -> k = kc + 16 + half*8 + c-8
// (two 16B runs) when the source is stored transposed as BT[n][k].
__device__ __forceinline__ v16bf load_b_bf16(const __bf16* colptr, int kc, int half) {
  const __bf16* p = colptr + kc + half * 8;
  v8bf lo = *(const v8bf*)p;
  v8bf hi = *(const v8bf*)(p + 16);
  v16bf b;
#pragma unroll
  for (int i = 0; i < 8; ++i) { b[i] = lo[i]; b[8 + i] = hi[i]; }
  return b;
}

__device__ __forceinline__ v8f wmma_bf16(v16bf a, v16bf b, v8f c) {
  // (neg_a, A, neg_b, B, c_mod, C, reuse_a, reuse_b)
  return __builtin_amdgcn_wmma_f32_16x16x32_bf16(false, a, false, b,
                                                 (short)0, c, false, false);
}

// ---------------------------------------------------------------------------
// Kernel 0: Hbf = bf16(x / 255), vectorized 8 elems/thread.
// ---------------------------------------------------------------------------
__global__ void __launch_bounds__(256)
k_prep_x(const float* __restrict__ x, __bf16* __restrict__ Hbf) {
  const size_t i = ((size_t)blockIdx.x * 256 + threadIdx.x) * 8;
  const float s = 1.0f / 255.0f;
  float4 f0 = *(const float4*)(x + i);
  float4 f1 = *(const float4*)(x + i + 4);
  v8bf h;
  h[0] = (__bf16)(f0.x * s); h[1] = (__bf16)(f0.y * s);
  h[2] = (__bf16)(f0.z * s); h[3] = (__bf16)(f0.w * s);
  h[4] = (__bf16)(f1.x * s); h[5] = (__bf16)(f1.y * s);
  h[6] = (__bf16)(f1.z * s); h[7] = (__bf16)(f1.w * s);
  *(v8bf*)(Hbf + i) = h;
}

// ---------------------------------------------------------------------------
// Kernel 1: weights -> bf16, transposed (WT[n][k] = W[k][n]).
// ---------------------------------------------------------------------------
__global__ void k_prep_weights(const float* __restrict__ Wq, const float* __restrict__ Wk,
                               const float* __restrict__ Wv, const float* __restrict__ Wp,
                               __bf16* WqT, __bf16* WkT, __bf16* WvT, __bf16* WpT) {
  int idx = blockIdx.x * blockDim.x + threadIdx.x;     // 0 .. 512*512-1
  int w   = blockIdx.y;
  const float* src = (w == 0) ? Wq : (w == 1) ? Wk : (w == 2) ? Wv : Wp;
  __bf16*      dst = (w == 0) ? WqT : (w == 1) ? WkT : (w == 2) ? WvT : WpT;
  int n = idx >> 9;
  int k = idx & 511;
  dst[idx] = (__bf16)src[k * C_DIM + n];
}

// ---------------------------------------------------------------------------
// Kernel 2: QKV projection. grid=(2048,3), block=256 (8 waves).
// 16x64 strip per wave; double-buffered fragments across the k-loop.
// V stored transposed [b][c][n] for the P@V B-fragments.
// ---------------------------------------------------------------------------
__global__ void __launch_bounds__(256)
k_qkv(const __bf16* __restrict__ Hbf,
      const __bf16* __restrict__ WqT, const __bf16* __restrict__ WkT,
      const __bf16* __restrict__ WvT,
      __bf16* __restrict__ Q, __bf16* __restrict__ K, __bf16* __restrict__ Vt) {
  const int tid  = threadIdx.x;
  const int wave = tid >> 5, lane = tid & 31;
  const int strip = blockIdx.x * 8 + wave;     // 0 .. 16383
  const int tm  = strip >> 3;                  // token tile   (0..2047)
  const int tng = strip & 7;                   // 64-ch group  (0..7)
  const int row = lane & 15, kg = lane >> 4;
  const int sel = blockIdx.y;                  // 0=Q 1=K 2=V

  const __bf16* WT = (sel == 0) ? WqT : (sel == 1) ? WkT : WvT;
  const __bf16* arow = Hbf + (size_t)(tm * 16 + row) * C_DIM;
  const __bf16* bcol[4];
#pragma unroll
  for (int j = 0; j < 4; ++j)
    bcol[j] = WT + (size_t)(tng * 64 + j * 16 + row) * C_DIM;

  // software pipeline: prologue loads for kc=0
  v16bf a  = load_a_bf16(arow, 0, kg);
  v16bf b0 = load_b_bf16(bcol[0], 0, kg);
  v16bf b1 = load_b_bf16(bcol[1], 0, kg);
  v16bf b2 = load_b_bf16(bcol[2], 0, kg);
  v16bf b3 = load_b_bf16(bcol[3], 0, kg);

  v8f acc[4] = {};
#pragma unroll 4
  for (int kc = 0; kc < C_DIM; kc += 32) {
    const int kn = (kc + 32) & (C_DIM - 1);    // wraps to 0 on last step
    v16bf an  = load_a_bf16(arow, kn, kg);     // next-step loads issued FIRST
    v16bf c0  = load_b_bf16(bcol[0], kn, kg);
    v16bf c1  = load_b_bf16(bcol[1], kn, kg);
    v16bf c2  = load_b_bf16(bcol[2], kn, kg);
    v16bf c3  = load_b_bf16(bcol[3], kn, kg);
    acc[0] = wmma_bf16(a, b0, acc[0]);         // WMMAs cover the load latency
    acc[1] = wmma_bf16(a, b1, acc[1]);
    acc[2] = wmma_bf16(a, b2, acc[2]);
    acc[3] = wmma_bf16(a, b3, acc[3]);
    a = an; b0 = c0; b1 = c1; b2 = c2; b3 = c3;
    SCHED_GROUP(0x020, 10);                    // 10 VMEM reads (2 A + 8 B) ...
    SCHED_GROUP(0x008, 4);                     // ... then 4 WMMAs
  }

  // C/D layout: lane comp r -> m = kg*8 + r, n = lane&15
  if (sel != 2) {
    __bf16* dst = (sel == 0) ? Q : K;
#pragma unroll
    for (int j = 0; j < 4; ++j)
#pragma unroll
      for (int r = 0; r < 8; ++r) {
        int m = tm * 16 + kg * 8 + r;
        dst[(size_t)m * C_DIM + tng * 64 + j * 16 + row] = (__bf16)acc[j][r];
      }
  } else {
#pragma unroll
    for (int j = 0; j < 4; ++j)
#pragma unroll
      for (int r = 0; r < 8; ++r) {
        int t  = tm * 16 + kg * 8 + r;
        int b  = t >> 12;            // batch
        int nl = t & (NTOK - 1);     // token within batch
        int ch = tng * 64 + j * 16 + row;
        Vt[((size_t)(b * C_DIM + ch)) * NTOK + nl] = (__bf16)acc[j][r];
      }
  }
}

// ---------------------------------------------------------------------------
// Kernel 3: attention. grid=(256 qtiles, 8 batches), block=256 (8 waves).
// Phase 1: S[16][4096] = Q_tile @ K^T into LDS.
// Phase 2: exact softmax stats; exp written back unnormalized (f32).
// Phase 3: O = P @ V; 1/rowsum folded into the f32 accumulator epilogue.
// LDS: 16*4096*4 = 256KB (CDNA5 allows up to 320KB per workgroup).
// ---------------------------------------------------------------------------
__global__ void __launch_bounds__(256)
k_attn(const __bf16* __restrict__ Q, const __bf16* __restrict__ K,
       const __bf16* __restrict__ Vt, __bf16* __restrict__ O) {
  extern __shared__ float smem[];
  float* S      = smem;               // 16 * 4096
  float* red    = S + 16 * NTOK;      // 256 partials
  float* rowmax = red + 256;          // 16
  float* rowrcp = rowmax + 16;        // 16

  const int b = blockIdx.y, qt = blockIdx.x;
  const int tid = threadIdx.x, wave = tid >> 5, lane = tid & 31;
  const int row = lane & 15, kg = lane >> 4;

  const __bf16* qrow = Q + (size_t)(b * NTOK + qt * 16 + row) * C_DIM;

  // ---- Phase 1: logits into LDS (4 key-tiles per A-fragment) --------------
  for (int ntc = 0; ntc < 8; ++ntc) {
    const int keyb = wave * 512 + ntc * 64;
    const __bf16* krow[4];
#pragma unroll
    for (int j = 0; j < 4; ++j)
      krow[j] = K + (size_t)(b * NTOK + keyb + j * 16 + row) * C_DIM;
    if (ntc + 1 < 8)  // warm L2/L0 for the next chunk of key rows
      __builtin_prefetch(K + (size_t)(b * NTOK + keyb + 64 + row) * C_DIM, 0, 1);

    v16bf a  = load_a_bf16(qrow, 0, kg);
    v16bf b0 = load_b_bf16(krow[0], 0, kg);
    v16bf b1 = load_b_bf16(krow[1], 0, kg);
    v16bf b2 = load_b_bf16(krow[2], 0, kg);
    v16bf b3 = load_b_bf16(krow[3], 0, kg);

    v8f acc[4] = {};
#pragma unroll 4
    for (int kc = 0; kc < C_DIM; kc += 32) {
      const int kn = (kc + 32) & (C_DIM - 1);
      v16bf an = load_a_bf16(qrow, kn, kg);
      v16bf c0 = load_b_bf16(krow[0], kn, kg);
      v16bf c1 = load_b_bf16(krow[1], kn, kg);
      v16bf c2 = load_b_bf16(krow[2], kn, kg);
      v16bf c3 = load_b_bf16(krow[3], kn, kg);
      acc[0] = wmma_bf16(a, b0, acc[0]);
      acc[1] = wmma_bf16(a, b1, acc[1]);
      acc[2] = wmma_bf16(a, b2, acc[2]);
      acc[3] = wmma_bf16(a, b3, acc[3]);
      a = an; b0 = c0; b1 = c1; b2 = c2; b3 = c3;
      SCHED_GROUP(0x020, 10);
      SCHED_GROUP(0x008, 4);
    }
#pragma unroll
    for (int j = 0; j < 4; ++j)
#pragma unroll
      for (int r = 0; r < 8; ++r)
        S[(kg * 8 + r) * NTOK + keyb + j * 16 + row] = acc[j][r];
  }
  __syncthreads();

  // ---- Phase 2: softmax stats (16 rows x 16 threads/row, 256 elems each) --
  {
    const int m = tid >> 4, j = tid & 15;
    float* Srow = S + m * NTOK + j * 256;
    float mx = -3.0e38f;
    for (int i = 0; i < 256; ++i) mx = fmaxf(mx, Srow[i]);
    red[m * 16 + j] = mx;
    __syncthreads();
    if (tid < 16) {
      float v = -3.0e38f;
      for (int jj = 0; jj < 16; ++jj) v = fmaxf(v, red[tid * 16 + jj]);
      rowmax[tid] = v;
    }
    __syncthreads();
    const float rm = rowmax[m];
    float sum = 0.0f;
    for (int i = 0; i < 256; ++i) {
      float e = __expf(Srow[i] - rm);
      Srow[i] = e;          // unnormalized numerators stay in LDS
      sum += e;
    }
    red[m * 16 + j] = sum;
    __syncthreads();
    if (tid < 16) {
      float v = 0.0f;
      for (int jj = 0; jj < 16; ++jj) v += red[tid * 16 + jj];
      rowrcp[tid] = 1.0f / v;
    }
    __syncthreads();
  }

  // ---- Phase 3: O = P @ V  (each wave owns 64 output channels) ------------
  const float* prow = S + row * NTOK;
  const __bf16* vcol[4];
#pragma unroll
  for (int nt = 0; nt < 4; ++nt)
    vcol[nt] = Vt + ((size_t)(b * C_DIM + wave * 64 + nt * 16 + row)) * NTOK;

  v16bf a  = load_a_f32(prow, 0, kg);
  v16bf b0 = load_b_bf16(vcol[0], 0, kg);
  v16bf b1 = load_b_bf16(vcol[1], 0, kg);
  v16bf b2 = load_b_bf16(vcol[2], 0, kg);
  v16bf b3 = load_b_bf16(vcol[3], 0, kg);

  v8f acc[4] = {};
#pragma unroll 4
  for (int kb = 0; kb < NTOK; kb += 32) {
    const int kn = (kb + 32) & (NTOK - 1);
    v16bf an = load_a_f32(prow, kn, kg);       // LDS reads + cvt
    v16bf c0 = load_b_bf16(vcol[0], kn, kg);   // global 32B reads
    v16bf c1 = load_b_bf16(vcol[1], kn, kg);
    v16bf c2 = load_b_bf16(vcol[2], kn, kg);
    v16bf c3 = load_b_bf16(vcol[3], kn, kg);
    acc[0] = wmma_bf16(a, b0, acc[0]);
    acc[1] = wmma_bf16(a, b1, acc[1]);
    acc[2] = wmma_bf16(a, b2, acc[2]);
    acc[3] = wmma_bf16(a, b3, acc[3]);
    a = an; b0 = c0; b1 = c1; b2 = c2; b3 = c3;
    SCHED_GROUP(0x100, 4);                     // 4 DS reads (P tile)
    SCHED_GROUP(0x020, 8);                     // 8 VMEM reads (V tiles)
    SCHED_GROUP(0x008, 4);                     // then 4 WMMAs
  }
  // Epilogue: fold 1/rowsum into the f32 accumulator (row of comp r = kg*8+r)
  float rcp8[8];
#pragma unroll
  for (int r = 0; r < 8; ++r) rcp8[r] = rowrcp[kg * 8 + r];
#pragma unroll
  for (int nt = 0; nt < 4; ++nt) {
#pragma unroll
    for (int r = 0; r < 8; ++r) {
      const int t  = b * NTOK + qt * 16 + kg * 8 + r;
      const int ch = wave * 64 + nt * 16 + row;
      O[(size_t)t * C_DIM + ch] = (__bf16)(acc[nt][r] * rcp8[r]);
    }
  }
}

// ---------------------------------------------------------------------------
// Kernel 4: out = x + O @ Wp. 16x64 strip per wave like QKV.
// ---------------------------------------------------------------------------
__global__ void __launch_bounds__(256)
k_proj(const float* __restrict__ x, const __bf16* __restrict__ O,
       const __bf16* __restrict__ WpT, float* __restrict__ out) {
  const int tid  = threadIdx.x;
  const int wave = tid >> 5, lane = tid & 31;
  const int strip = blockIdx.x * 8 + wave;
  const int tm = strip >> 3, tng = strip & 7;
  const int row = lane & 15, kg = lane >> 4;

  const __bf16* arow = O + (size_t)(tm * 16 + row) * C_DIM;
  const __bf16* bcol[4];
#pragma unroll
  for (int j = 0; j < 4; ++j)
    bcol[j] = WpT + (size_t)(tng * 64 + j * 16 + row) * C_DIM;

  v16bf a  = load_a_bf16(arow, 0, kg);
  v16bf b0 = load_b_bf16(bcol[0], 0, kg);
  v16bf b1 = load_b_bf16(bcol[1], 0, kg);
  v16bf b2 = load_b_bf16(bcol[2], 0, kg);
  v16bf b3 = load_b_bf16(bcol[3], 0, kg);

  v8f acc[4] = {};
#pragma unroll 4
  for (int kc = 0; kc < C_DIM; kc += 32) {
    const int kn = (kc + 32) & (C_DIM - 1);
    v16bf an = load_a_bf16(arow, kn, kg);
    v16bf c0 = load_b_bf16(bcol[0], kn, kg);
    v16bf c1 = load_b_bf16(bcol[1], kn, kg);
    v16bf c2 = load_b_bf16(bcol[2], kn, kg);
    v16bf c3 = load_b_bf16(bcol[3], kn, kg);
    acc[0] = wmma_bf16(a, b0, acc[0]);
    acc[1] = wmma_bf16(a, b1, acc[1]);
    acc[2] = wmma_bf16(a, b2, acc[2]);
    acc[3] = wmma_bf16(a, b3, acc[3]);
    a = an; b0 = c0; b1 = c1; b2 = c2; b3 = c3;
    SCHED_GROUP(0x020, 10);
    SCHED_GROUP(0x008, 4);
  }
#pragma unroll
  for (int j = 0; j < 4; ++j)
#pragma unroll
    for (int r = 0; r < 8; ++r) {
      size_t idx = (size_t)(tm * 16 + kg * 8 + r) * C_DIM + tng * 64 + j * 16 + row;
      out[idx] = x[idx] + acc[j][r];     // residual
    }
}

// ---------------------------------------------------------------------------
extern "C" void kernel_launch(void* const* d_in, const int* in_sizes, int n_in,
                              void* d_out, int out_size, void* d_ws, size_t ws_size,
                              hipStream_t stream) {
  (void)in_sizes; (void)n_in; (void)out_size; (void)ws_size;
  const float* x  = (const float*)d_in[0];
  const float* Wq = (const float*)d_in[1];
  const float* Wk = (const float*)d_in[2];
  const float* Wv = (const float*)d_in[3];
  const float* Wp = (const float*)d_in[4];
  float* out = (float*)d_out;

  // Workspace: 4 transposed bf16 weights (0.5MB each) + Hbf, Q, K, Vt, O
  // (32MB each bf16)  => ~162MB.
  char* ws = (char*)d_ws;
  size_t off = 0;
  auto carve = [&](size_t bytes) -> char* {
    char* p = ws + off;
    off += (bytes + 255) & ~(size_t)255;
    return p;
  };
  const size_t wbytes = (size_t)C_DIM * C_DIM * sizeof(__bf16);
  const size_t tbytes = (size_t)TOKENS * C_DIM * sizeof(__bf16);
  __bf16* WqT = (__bf16*)carve(wbytes);
  __bf16* WkT = (__bf16*)carve(wbytes);
  __bf16* WvT = (__bf16*)carve(wbytes);
  __bf16* WpT = (__bf16*)carve(wbytes);
  __bf16* Hbf = (__bf16*)carve(tbytes);
  __bf16* Qb  = (__bf16*)carve(tbytes);
  __bf16* Kb  = (__bf16*)carve(tbytes);
  __bf16* Vt  = (__bf16*)carve(tbytes);
  __bf16* Ob  = (__bf16*)carve(tbytes);

  // 0) x -> bf16(x/255), once
  k_prep_x<<<dim3(TOKENS * C_DIM / (256 * 8)), 256, 0, stream>>>(x, Hbf);

  // 1) weights -> bf16 transposed
  k_prep_weights<<<dim3((C_DIM * C_DIM) / 256, 4), 256, 0, stream>>>(
      Wq, Wk, Wv, Wp, WqT, WkT, WvT, WpT);

  // 2) QKV projection (16384 strips of 16x64 per matrix, 8 waves/block)
  k_qkv<<<dim3(2048, 3), 256, 0, stream>>>(Hbf, WqT, WkT, WvT, Qb, Kb, Vt);

  // 3) attention: one workgroup per (16-query tile, batch); 256KB+ LDS
  const size_t smem_bytes = (size_t)(16 * NTOK + 256 + 16 + 16) * sizeof(float);
  k_attn<<<dim3(NTOK / 16, BATCH), 256, smem_bytes, stream>>>(Qb, Kb, Vt, Ob);

  // 4) projection + residual
  k_proj<<<dim3(2048, 1), 256, 0, stream>>>(x, Ob, WpT, out);
}
